// MHAttention_46961172414945
// MI455X (gfx1250) — matrix operand order repaired and checked by reading, not compiled, not run on hardware
//
#include <hip/hip_runtime.h>
#include <hip/hip_bf16.h>

typedef _Float16 half8 __attribute__((ext_vector_type(8)));
typedef _Float16 v16h  __attribute__((ext_vector_type(16)));
typedef float    v8f   __attribute__((ext_vector_type(8)));
typedef float    vf4   __attribute__((ext_vector_type(4)));

union V16u { v16h v; half8 h[2]; };

namespace cfg {
constexpr int B = 4, S = 2048, D = 512, H = 8, DK = 64;
constexpr int M = B * S;            // 8192 token rows
}

__device__ __forceinline__ v8f wmma_f16f32(v16h a, v16h b, v8f c) {
  // v_wmma_f32_16x16x32_f16: D = A(16x32) * B(32x16) + C(16x16 f32)
  return __builtin_amdgcn_wmma_f32_16x16x32_f16(
      /*neg_a=*/false, a, /*neg_b=*/false, b,
      /*c_mod=*/(short)0, c, /*reuse_a=*/false, /*reuse_b=*/false);
}

// ---- DPP16 butterfly max within each 16-lane row --------------------------
// 0xB1 = quad_perm(1,0,3,2) xor1 ; 0x4E = quad_perm(2,3,0,1) xor2
// 0x141 = row_half_mirror xor4   ; 0x140 = row_mirror xor8
template <int CTRL>
__device__ __forceinline__ float dpp_xor(float x) {
  int v = __builtin_amdgcn_update_dpp(0, __float_as_int(x), CTRL, 0xf, 0xf, true);
  return __int_as_float(v);
}
__device__ __forceinline__ float row16_max(float x) {
  x = fmaxf(x, dpp_xor<0xB1>(x));
  x = fmaxf(x, dpp_xor<0x4E>(x));
  x = fmaxf(x, dpp_xor<0x141>(x));
  x = fmaxf(x, dpp_xor<0x140>(x));
  return x;
}

// ---- operand loaders ------------------------------------------------------
// A operand (16x32 f16), ISA layout: M = lane%16; half h -> K = 16*(h/8) + 8*lh + h%8
__device__ __forceinline__ v16h load_a_f16(const _Float16* rowBase, int lh) {
  const half8* p = (const half8*)rowBase;          // row of >=32 halfs
  V16u u; u.h[0] = p[lh]; u.h[1] = p[2 + lh];      // K: 8lh..8lh+7, 16+8lh..
  return u.v;
}
__device__ __forceinline__ v16h load_a_f32cvt(const float* rowBase, int lh) {
  const vf4* p = (const vf4*)rowBase;              // row of >=32 floats
  vf4 x0 = p[2 * lh], x1 = p[2 * lh + 1];
  vf4 x2 = p[4 + 2 * lh], x3 = p[4 + 2 * lh + 1];
  V16u u;
#pragma unroll
  for (int i = 0; i < 4; ++i) {
    u.v[i]      = (_Float16)x0[i];
    u.v[4 + i]  = (_Float16)x1[i];
    u.v[8 + i]  = (_Float16)x2[i];
    u.v[12 + i] = (_Float16)x3[i];
  }
  return u.v;
}
// B operand (32x16 f16), ISA layout: N = lane%16; half h -> K = 16*(h/8) + 8*lh + h%8
__device__ __forceinline__ v16h load_b_f16(const _Float16* rowBase, int lh) {
  const half8* p = (const half8*)rowBase;          // "row" = column n of B, contiguous in K
  V16u u; u.h[0] = p[lh]; u.h[1] = p[2 + lh];
  return u.v;
}

// ---- prep kernels ---------------------------------------------------------
__global__ void prep_weights_kernel(const float* __restrict__ Wq, const float* __restrict__ Wk,
                                    const float* __restrict__ Wv, const float* __restrict__ Wo,
                                    _Float16* __restrict__ WqT, _Float16* __restrict__ WkT,
                                    _Float16* __restrict__ WvT, _Float16* __restrict__ WoT) {
  int idx = blockIdx.x * blockDim.x + threadIdx.x;   // 4 * 512*512 threads
  int which = idx >> 18;
  int e = idx & 262143;
  int n = e >> 9;          // transposed row = output column
  int k = e & 511;
  const float* W = (which == 0) ? Wq : (which == 1) ? Wk : (which == 2) ? Wv : Wo;
  _Float16*    T = (which == 0) ? WqT : (which == 1) ? WkT : (which == 2) ? WvT : WoT;
  T[(size_t)n * 512 + k] = (_Float16)W[(size_t)k * 512 + n];
}

__global__ void prep_mask_kernel(const int* __restrict__ mask, float* __restrict__ madd, int n) {
  int i = blockIdx.x * blockDim.x + threadIdx.x;
  if (i < n) madd[i] = 1.0e9f * ((float)mask[i] - 1.0f);
}

// ---- WMMA GEMM: Y = X @ W + bias, wave computes 16(M) x 64(N) -------------
// MODE 0: Q -> Qh (B,H,S,64) f16    MODE 1: K -> Kh (B,H,S,64) f16
// MODE 2: V -> Vth (B,H,64,S) f16   MODE 3: ctx(f16) @ Wo -> out f32 (B,S,D)
template <int MODE>
__global__ void proj_kernel(const float* __restrict__ Af32, const _Float16* __restrict__ Af16,
                            const _Float16* __restrict__ BT, const float* __restrict__ bias,
                            _Float16* __restrict__ outH, float* __restrict__ outF) {
  const int lane = threadIdx.x;        // 0..31
  const int wave = threadIdx.y;        // 0..7
  const int lh = lane >> 4, ln = lane & 15;
  const int m0 = blockIdx.x * 128 + wave * 16;
  const int n0 = blockIdx.y * 64;
  const int row = m0 + ln;

  v8f acc[4] = {};
  const float*    arow32 = (MODE < 3) ? (Af32 + (size_t)row * cfg::D) : nullptr;
  const _Float16* arow16 = (MODE == 3) ? (Af16 + (size_t)row * cfg::D) : nullptr;

  for (int k0 = 0; k0 < cfg::D; k0 += 32) {
    v16h a = (MODE < 3) ? load_a_f32cvt(arow32 + k0, lh) : load_a_f16(arow16 + k0, lh);
#pragma unroll
    for (int t = 0; t < 4; ++t) {
      const _Float16* bcol = BT + (size_t)(n0 + t * 16 + ln) * cfg::D + k0;
      v16h b = load_b_f16(bcol, lh);
      acc[t] = wmma_f16f32(a, b, acc[t]);
    }
  }

#pragma unroll
  for (int t = 0; t < 4; ++t) {
    int n = n0 + t * 16 + ln;
    float bv = bias[n];
#pragma unroll
    for (int r = 0; r < 8; ++r) {
      int mrow = m0 + r + 8 * lh;               // C layout: M = r + 8*(lane/16), N = lane%16
      float v = acc[t][r] + bv;
      int b = mrow >> 11;                       // / S
      int s = mrow & (cfg::S - 1);
      if (MODE == 0 || MODE == 1) {
        int h = n >> 6, dk = n & 63;
        outH[(((size_t)(b * cfg::H + h) * cfg::S + s) << 6) + dk] = (_Float16)v;
      } else if (MODE == 2) {
        int h = n >> 6, dk = n & 63;
        outH[((size_t)(b * cfg::H + h) * cfg::DK + dk) * cfg::S + s] = (_Float16)v;
      } else {
        outF[(size_t)mrow * cfg::D + n] = v;
      }
    }
  }
}

// ---- flash attention: 1 wave = 1 (b,h,16-query tile) ----------------------
// Softmax denominators accumulated by an extra WMMA against an all-ones B.
__global__ void flash_kernel(const _Float16* __restrict__ Qh, const _Float16* __restrict__ Kh,
                             const _Float16* __restrict__ Vth, const float* __restrict__ maskAdd,
                             _Float16* __restrict__ Ctxh) {
  __shared__ __align__(16) _Float16 ldsP[2][16][32];   // double-buffered P tile
  const int lane = threadIdx.x, lh = lane >> 4, ln = lane & 15;
  const int q0 = blockIdx.x * 16;
  const int h = blockIdx.y, b = blockIdx.z;
  const int bh = b * cfg::H + h;

  const _Float16* qrow = Qh + ((size_t)bh * cfg::S + q0 + ln) * cfg::DK;
  v16h qa0 = load_a_f16(qrow, lh);        // dk 0..31
  v16h qa1 = load_a_f16(qrow + 32, lh);   // dk 32..63

  v16h ones;
  {
    V16u u;
#pragma unroll
    for (int i = 0; i < 16; ++i) u.v[i] = (_Float16)1.0f;
    ones = u.v;
  }

  float mrow[8];
  v8f acc[4] = {};
  v8f accl = {};
#pragma unroll
  for (int r = 0; r < 8; ++r) mrow[r] = -3.0e38f;

  const float scale = 0.125f;             // 1/sqrt(64)
  const float* mb = maskAdd + (size_t)b * cfg::S;

  for (int kc = 0; kc < cfg::S; kc += 32) {
    const int buf = (kc >> 5) & 1;
    // scores: two 16x16 tiles over keys [kc, kc+16) and [kc+16, kc+32)
    v8f c0 = {}, c1 = {};
    {
      const _Float16* kr0 = Kh + ((size_t)bh * cfg::S + kc + ln) * cfg::DK;
      const _Float16* kr1 = Kh + ((size_t)bh * cfg::S + kc + 16 + ln) * cfg::DK;
      c0 = wmma_f16f32(qa0, load_b_f16(kr0, lh), c0);
      c0 = wmma_f16f32(qa1, load_b_f16(kr0 + 32, lh), c0);
      c1 = wmma_f16f32(qa0, load_b_f16(kr1, lh), c1);
      c1 = wmma_f16f32(qa1, load_b_f16(kr1 + 32, lh), c1);
    }
    float mk0 = mb[kc + ln], mk1 = mb[kc + 16 + ln];

    float s0[8], s1[8], mnew[8], alpha[8];
#pragma unroll
    for (int r = 0; r < 8; ++r) {
      s0[r] = c0[r] * scale + mk0;
      s1[r] = c1[r] * scale + mk1;
      float cm = row16_max(fmaxf(s0[r], s1[r]));
      mnew[r] = fmaxf(mrow[r], cm);
    }
#pragma unroll
    for (int r = 0; r < 8; ++r) {
      alpha[r] = __expf(mrow[r] - mnew[r]);
      float p0 = __expf(s0[r] - mnew[r]);
      float p1 = __expf(s1[r] - mnew[r]);
      ldsP[buf][r + 8 * lh][ln] = (_Float16)p0;        // P[query][key_local]
      ldsP[buf][r + 8 * lh][16 + ln] = (_Float16)p1;
      mrow[r] = mnew[r];
    }
#pragma unroll
    for (int t = 0; t < 4; ++t)
#pragma unroll
      for (int r = 0; r < 8; ++r) acc[t][r] = acc[t][r] * alpha[r];
#pragma unroll
    for (int r = 0; r < 8; ++r) accl[r] = accl[r] * alpha[r];

    __syncthreads();   // order ds stores before cross-lane ds loads
    v16h pa = load_a_f16(&ldsP[buf][ln][0], lh);        // P as A operand (16x32)

    // rowsum(P) via WMMA against all-ones B: l += P x 1
    accl = wmma_f16f32(pa, ones, accl);
#pragma unroll
    for (int t = 0; t < 4; ++t) {
      const _Float16* vcol = Vth + ((size_t)bh * cfg::DK + t * 16 + ln) * cfg::S + kc;
      acc[t] = wmma_f16f32(pa, load_b_f16(vcol, lh), acc[t]);
    }
  }

#pragma unroll
  for (int t = 0; t < 4; ++t)
#pragma unroll
    for (int r = 0; r < 8; ++r) {
      float v = acc[t][r] / accl[r];
      int s = q0 + r + 8 * lh;
      Ctxh[((size_t)b * cfg::S + s) * cfg::D + h * cfg::DK + t * 16 + ln] = (_Float16)v;
    }
}

// ---- host launch ----------------------------------------------------------
extern "C" void kernel_launch(void* const* d_in, const int* in_sizes, int n_in,
                              void* d_out, int out_size, void* d_ws, size_t ws_size,
                              hipStream_t stream) {
  const float* query = (const float*)d_in[0];
  const float* value = (const float*)d_in[1];
  const int*   amask = (const int*)d_in[2];
  const float* Wq = (const float*)d_in[3]; const float* bq = (const float*)d_in[4];
  const float* Wk = (const float*)d_in[5]; const float* bk = (const float*)d_in[6];
  const float* Wv = (const float*)d_in[7]; const float* bv = (const float*)d_in[8];
  const float* Wo = (const float*)d_in[9]; const float* bo = (const float*)d_in[10];
  float* out = (float*)d_out;

  char* ws = (char*)d_ws;
  size_t off = 0;
  auto alloc = [&](size_t bytes) -> void* {
    void* p = ws + off; off += (bytes + 255) & ~(size_t)255; return p;
  };
  const size_t nTok = (size_t)cfg::B * cfg::S;                 // 8192
  _Float16* Qh   = (_Float16*)alloc(nTok * cfg::D * 2);        // (B,H,S,64)
  _Float16* Kh   = (_Float16*)alloc(nTok * cfg::D * 2);        // (B,H,S,64)
  _Float16* Vth  = (_Float16*)alloc(nTok * cfg::D * 2);        // (B,H,64,S)
  _Float16* Ctxh = (_Float16*)alloc(nTok * cfg::D * 2);        // (B,S,512)
  _Float16* WqT  = (_Float16*)alloc((size_t)512 * 512 * 2);
  _Float16* WkT  = (_Float16*)alloc((size_t)512 * 512 * 2);
  _Float16* WvT  = (_Float16*)alloc((size_t)512 * 512 * 2);
  _Float16* WoT  = (_Float16*)alloc((size_t)512 * 512 * 2);
  float*    madd = (float*)alloc(nTok * 4);

  prep_weights_kernel<<<4096, 256, 0, stream>>>(Wq, Wk, Wv, Wo, WqT, WkT, WvT, WoT);
  prep_mask_kernel<<<32, 256, 0, stream>>>(amask, madd, (int)nTok);

  dim3 gb(cfg::M / 128, cfg::D / 64), tb(32, 8);
  proj_kernel<0><<<gb, tb, 0, stream>>>(query, nullptr, WqT, bq, Qh, nullptr);
  proj_kernel<1><<<gb, tb, 0, stream>>>(value, nullptr, WkT, bk, Kh, nullptr);
  proj_kernel<2><<<gb, tb, 0, stream>>>(value, nullptr, WvT, bv, Vth, nullptr);

  flash_kernel<<<dim3(cfg::S / 16, cfg::H, cfg::B), 32, 0, stream>>>(Qh, Kh, Vth, madd, Ctxh);

  proj_kernel<3><<<gb, tb, 0, stream>>>(nullptr, Ctxh, WoT, bo, nullptr, out);
}
